// CrossWOZTRADE_23424751633071
// MI455X (gfx1250) — matrix operand order repaired and checked by reading, not compiled, not run on hardware
//
#include <hip/hip_runtime.h>
#include <hip/hip_bf16.h>
#include <math.h>

// ---- model dims ----
#define V_     20000
#define H_     400
#define B_     16
#define S_     256
#define NSLOT_ 30
#define T_     10
#define G_     3
#define H3_    1200
#define NB_    480          // NSLOT_*B_
#define KP_    416          // K padded to 13*32 for WMMA bf16 K-loop
#define KT_    13           // KP_/32

typedef __attribute__((ext_vector_type(16))) __bf16 v16bf;
typedef __attribute__((ext_vector_type(8)))  float  v8f;

__device__ __forceinline__ float sigm(float x) { return 1.0f / (1.0f + __expf(-x)); }

// ============================================================================
// Generic bf16 WMMA GEMM:  C(M,N) = A(M,KP) * B(N,KP)^T (+ bias[n])
// Fragment layouts per CDNA5 ISA 7.12.2 (see round-1 notes).
// ============================================================================
template <int NT>
__global__ void k_gemm_bf16(const __bf16* __restrict__ A, int lda,
                            const __bf16* __restrict__ Bm, int ldb,
                            float* __restrict__ C, int ldc,
                            const float* __restrict__ bias) {
    const int m0   = blockIdx.y * 16;
    const int n0   = blockIdx.x * 16 * NT;
    const int lane = threadIdx.x;
    const int q    = lane & 15;
    const int half = lane >> 4;

    v8f acc[NT];
#pragma unroll
    for (int i = 0; i < NT; ++i) acc[i] = (v8f){0,0,0,0,0,0,0,0};

    const __bf16* Ap = A + (size_t)(m0 + q) * lda + half * 8;

    for (int kt = 0; kt < KT_; ++kt) {
        const int k0 = kt * 32;
        v16bf a;
        *((uint4*)&a)       = *((const uint4*)(Ap + k0));
        *(((uint4*)&a) + 1) = *((const uint4*)(Ap + k0 + 16));
#pragma unroll
        for (int i = 0; i < NT; ++i) {
            const __bf16* Bp = Bm + (size_t)(n0 + i * 16 + q) * ldb + k0 + half * 8;
            v16bf b;
            *((uint4*)&b)       = *((const uint4*)(Bp));
            *(((uint4*)&b) + 1) = *((const uint4*)(Bp + 16));
            acc[i] = __builtin_amdgcn_wmma_f32_16x16x32_bf16(
                false, a, false, b, (short)0, acc[i], false, false);
        }
    }

#pragma unroll
    for (int i = 0; i < NT; ++i) {
        const int ncol = n0 + i * 16 + q;
        const float bv = bias ? bias[ncol] : 0.0f;
#pragma unroll
        for (int v = 0; v < 8; ++v) {
            const int m = m0 + half * 8 + v;
            C[(size_t)m * ldc + ncol] = acc[i][v] + bv;
        }
    }
}

// ============================================================================
// Fused bi-GRU encoder scan: ONE launch, grid = 2 (direction), block = 1024.
// h state: fp32 in registers (fixed tid->(b,j) map), bf16 copy in LDS as the
// WMMA A operand. Per step: 25 waves x 3 tiles compute gh = h@Whh^T + bhh
// into LDS, barrier, GRU gate + masked atomic accumulate into enc_acc.
// ============================================================================
__global__ __launch_bounds__(1024)
void k_enc_scan(const float* __restrict__ gi_f, const float* __restrict__ gi_b,
                const __bf16* __restrict__ Whhf, const __bf16* __restrict__ Whhb,
                const float* __restrict__ bhhf, const float* __restrict__ bhhb,
                const int* __restrict__ lens,
                float* __restrict__ enc_acc, float* __restrict__ h_out) {
    __shared__ float  sh_gh[B_ * H3_];    // 76800 B
    __shared__ __bf16 sh_hbf[B_ * KP_];   // 13312 B

    const int dir = blockIdx.x;
    const float*  gi  = dir ? gi_b : gi_f;
    const __bf16* Whh = dir ? Whhb : Whhf;
    const float*  bhh = dir ? bhhb : bhhf;

    const int tid  = threadIdx.x;
    const int wave = tid >> 5;
    const int lane = tid & 31;
    const int q    = lane & 15;
    const int half = lane >> 4;

    for (int i = tid; i < B_ * KP_; i += 1024) sh_hbf[i] = (__bf16)0.0f;
    float hreg[7];
#pragma unroll
    for (int e = 0; e < 7; ++e) hreg[e] = 0.0f;
    __syncthreads();

    for (int step = 0; step < S_; ++step) {
        const int s = dir ? (S_ - 1 - step) : step;

        // ---- phase 1: gh = h @ Whh^T + bhh (WMMA bf16, waves 0..24) ----
        if (wave < 25) {
            const int n0 = wave * 48;
            v8f a0 = (v8f){0,0,0,0,0,0,0,0}, a1 = a0, a2 = a0;
            const __bf16* hp = &sh_hbf[q * KP_ + half * 8];
            for (int kt = 0; kt < KT_; ++kt) {
                const int k0 = kt * 32;
                v16bf a;
                *((uint4*)&a)       = *((const uint4*)(hp + k0));
                *(((uint4*)&a) + 1) = *((const uint4*)(hp + k0 + 16));
                const __bf16* B0 = Whh + (size_t)(n0 + q) * KP_ + k0 + half * 8;
                v16bf b0, b1, b2;
                *((uint4*)&b0)       = *((const uint4*)(B0));
                *(((uint4*)&b0) + 1) = *((const uint4*)(B0 + 16));
                *((uint4*)&b1)       = *((const uint4*)(B0 + 16 * KP_));
                *(((uint4*)&b1) + 1) = *((const uint4*)(B0 + 16 * KP_ + 16));
                *((uint4*)&b2)       = *((const uint4*)(B0 + 32 * KP_));
                *(((uint4*)&b2) + 1) = *((const uint4*)(B0 + 32 * KP_ + 16));
                a0 = __builtin_amdgcn_wmma_f32_16x16x32_bf16(false, a, false, b0, (short)0, a0, false, false);
                a1 = __builtin_amdgcn_wmma_f32_16x16x32_bf16(false, a, false, b1, (short)0, a1, false, false);
                a2 = __builtin_amdgcn_wmma_f32_16x16x32_bf16(false, a, false, b2, (short)0, a2, false, false);
            }
#pragma unroll
            for (int v = 0; v < 8; ++v) {
                const int m = half * 8 + v;          // batch row 0..15
                sh_gh[m * H3_ + n0 + q]      = a0[v] + bhh[n0 + q];
                sh_gh[m * H3_ + n0 + 16 + q] = a1[v] + bhh[n0 + 16 + q];
                sh_gh[m * H3_ + n0 + 32 + q] = a2[v] + bhh[n0 + 32 + q];
            }
        }
        __syncthreads();

        // ---- phase 2: GRU gate + mask + output accumulation ----
#pragma unroll
        for (int e = 0; e < 7; ++e) {
            const int idx = tid + e * 1024;
            if (idx < B_ * H_) {
                const int b = idx / H_, j = idx % H_;
                const float* gir = gi + ((size_t)s * B_ + b) * H3_;
                const float r  = sigm(gir[j] + sh_gh[b * H3_ + j]);
                const float z  = sigm(gir[H_ + j] + sh_gh[b * H3_ + H_ + j]);
                const float nn = tanhf(gir[2 * H_ + j] + r * sh_gh[b * H3_ + 2 * H_ + j]);
                const float hold = hreg[e];
                const float hnew = (1.0f - z) * nn + z * hold;
                const bool  mt   = s < lens[b];
                const float hk   = mt ? hnew : hold;
                hreg[e] = hk;
                sh_hbf[b * KP_ + j] = (__bf16)hk;
                if (mt) atomicAdd(&enc_acc[((size_t)b * S_ + s) * H_ + j], hnew);
            }
        }
        __syncthreads();
    }

#pragma unroll
    for (int e = 0; e < 7; ++e) {
        const int idx = tid + e * 1024;
        if (idx < B_ * H_) h_out[(size_t)dir * B_ * H_ + idx] = hreg[e];
    }
}

// ---- f32 -> bf16 cast with K padding (rows x H_) -> (rows x KP_) ----
__global__ void k_cast_pad(const float* __restrict__ src, __bf16* __restrict__ dst,
                           int rows) {
    size_t idx = (size_t)blockIdx.x * blockDim.x + threadIdx.x;
    size_t total = (size_t)rows * KP_;
    if (idx >= total) return;
    int j = (int)(idx % KP_);
    size_t r = idx / KP_;
    dst[idx] = (j < H_) ? (__bf16)src[r * H_ + j] : (__bf16)0.0f;
}

// ---- embed story: x[s,b,:] = emb[story[b,s]] as padded bf16 ----
__global__ void k_embed_story(const int* __restrict__ story,
                              const float* __restrict__ emb,
                              __bf16* __restrict__ xbf) {
    size_t idx = (size_t)blockIdx.x * blockDim.x + threadIdx.x;
    size_t total = (size_t)S_ * B_ * KP_;
    if (idx >= total) return;
    int j = (int)(idx % KP_);
    size_t row = idx / KP_;           // row = s*B + b
    int s = (int)(row / B_), b = (int)(row % B_);
    int tok = story[b * S_ + s];
    xbf[idx] = (j < H_) ? (__bf16)emb[(size_t)tok * H_ + j] : (__bf16)0.0f;
}

// ---- decoder inputs: t==0 slot emb sum, else teacher-forced token emb ----
__global__ void k_dec_embed(const int* __restrict__ tgt, const int* __restrict__ dom,
                            const int* __restrict__ slo, const float* __restrict__ emb,
                            const float* __restrict__ slot_table,
                            float* __restrict__ xf, __bf16* __restrict__ xbf) {
    size_t idx = (size_t)blockIdx.x * blockDim.x + threadIdx.x;
    size_t total = (size_t)T_ * NB_ * KP_;
    if (idx >= total) return;
    int j = (int)(idx % KP_);
    size_t r = idx / KP_;             // r = t*NB + n
    int t = (int)(r / NB_), n = (int)(r % NB_);
    int slot = n / B_, b = n % B_;
    float v = 0.0f;
    if (j < H_) {
        if (t == 0) {
            v = slot_table[dom[slot] * H_ + j] + slot_table[slo[slot] * H_ + j];
        } else {
            int tok = tgt[((size_t)b * NSLOT_ + slot) * T_ + (t - 1)];
            v = emb[(size_t)tok * H_ + j];
        }
        xf[r * H_ + j] = v;
    }
    xbf[idx] = (__bf16)v;
}

// ---- decoder GRU gate (no mask) ----
__global__ void k_dec_gate(const float* __restrict__ gi, const float* __restrict__ gh,
                           float* __restrict__ h, __bf16* __restrict__ hbf, int t) {
    int idx = blockIdx.x * blockDim.x + threadIdx.x;
    if (idx >= NB_ * H_) return;
    int n = idx / H_, j = idx % H_;
    const float* gir = gi + ((size_t)t * NB_ + n) * H3_;
    const float* ghr = gh + (size_t)n * H3_;
    float r  = sigm(gir[j] + ghr[j]);
    float z  = sigm(gir[H_ + j] + ghr[H_ + j]);
    float nn = tanhf(gir[2 * H_ + j] + r * ghr[2 * H_ + j]);
    float hnew = (1.0f - z) * nn + z * h[idx];
    h[idx] = hnew;
    hbf[(size_t)n * KP_ + j] = (__bf16)hnew;
}

__global__ void k_add(const float* __restrict__ a, const float* __restrict__ b,
                      float* __restrict__ c, int nel) {
    int i = blockIdx.x * blockDim.x + threadIdx.x;
    if (i < nel) c[i] = a[i] + b[i];
}

// ---- decoder hidden init: tile(enc_hidden, nslot) ----
__global__ void k_hdec_init(const float* __restrict__ ench, float* __restrict__ h,
                            __bf16* __restrict__ hbf) {
    size_t idx = (size_t)blockIdx.x * blockDim.x + threadIdx.x;
    if (idx >= (size_t)NB_ * KP_) return;
    int n = (int)(idx / KP_), j = (int)(idx % KP_), b = n % B_;
    float v = (j < H_) ? ench[b * H_ + j] : 0.0f;
    if (j < H_) h[(size_t)n * H_ + j] = v;
    hbf[idx] = (__bf16)v;
}

// ---- attention scores: sc[n,s] = h[n] . enc[b= n%B, s] ----
__global__ void k_scores(const float* __restrict__ h, const float* __restrict__ enc,
                         float* __restrict__ sc) {
    int idx = blockIdx.x * blockDim.x + threadIdx.x;
    if (idx >= NB_ * S_) return;
    int n = idx / S_, s = idx % S_, b = n % B_;
    const float4* hp = (const float4*)(h + (size_t)n * H_);
    const float4* ep = (const float4*)(enc + ((size_t)b * S_ + s) * H_);
    float a = 0.0f;
#pragma unroll 4
    for (int k = 0; k < H_ / 4; ++k) {
        float4 x = hp[k], e = ep[k];
        a += x.x * e.x + x.y * e.y + x.z * e.z + x.w * e.w;
    }
    sc[idx] = a;
}

// ---- masked softmax over S, in place. grid = NB_, block = S_ ----
__global__ void k_attn_softmax(float* __restrict__ sc, const int* __restrict__ lens) {
    __shared__ float red[S_];
    int n = blockIdx.x, s = threadIdx.x, b = n % B_;
    bool valid = s < lens[b];
    float v = valid ? sc[(size_t)n * S_ + s] : -INFINITY;
    red[s] = v; __syncthreads();
    for (int off = S_ / 2; off > 0; off >>= 1) {
        if (s < off) red[s] = fmaxf(red[s], red[s + off]);
        __syncthreads();
    }
    float m = red[0]; __syncthreads();
    float e = valid ? __expf(v - m) : 0.0f;
    red[s] = e; __syncthreads();
    for (int off = S_ / 2; off > 0; off >>= 1) {
        if (s < off) red[s] += red[s + off];
        __syncthreads();
    }
    sc[(size_t)n * S_ + s] = e / red[0];
}

// ---- context[n] = sum_s prob[n,s] * enc[b,s]. grid = NB_, block = 256 ----
__global__ void k_context(const float* __restrict__ prob, const float* __restrict__ enc,
                          float* __restrict__ ctx) {
    __shared__ float p[S_];
    int n = blockIdx.x, b = n % B_;
    p[threadIdx.x] = prob[(size_t)n * S_ + threadIdx.x];
    __syncthreads();
    for (int j = threadIdx.x; j < H_; j += blockDim.x) {
        float a = 0.0f;
        for (int s = 0; s < S_; ++s)
            a += p[s] * enc[((size_t)b * S_ + s) * H_ + j];
        ctx[(size_t)n * H_ + j] = a;
    }
}

// ---- sw[n] = sigmoid(Wr . [h; context; x_in] + br). grid = NB_, block = 256 ----
__global__ void k_sw(const float* __restrict__ h, const float* __restrict__ ctx,
                     const float* __restrict__ xin, const float* __restrict__ Wr,
                     const float* __restrict__ br, float* __restrict__ sw, int t) {
    __shared__ float red[256];
    int n = blockIdx.x, tid = threadIdx.x;
    float a = 0.0f;
    for (int c = tid; c < H3_; c += 256) {
        float v;
        if      (c < H_)     v = h[(size_t)n * H_ + c];
        else if (c < 2 * H_) v = ctx[(size_t)n * H_ + (c - H_)];
        else                 v = xin[((size_t)t * NB_ + n) * H_ + (c - 2 * H_)];
        a += Wr[c] * v;
    }
    red[tid] = a; __syncthreads();
    for (int off = 128; off > 0; off >>= 1) {
        if (tid < off) red[tid] += red[tid + off];
        __syncthreads();
    }
    if (tid == 0) sw[n] = 1.0f / (1.0f + __expf(-(red[0] + br[0])));
}

// ---- gates (t==0 only) ----
__global__ void k_gates(const float* __restrict__ ctx, const float* __restrict__ Wg,
                        const float* __restrict__ bg, float* __restrict__ outg) {
    int idx = blockIdx.x * blockDim.x + threadIdx.x;
    if (idx >= NB_ * G_) return;
    int n = idx / G_, g = idx % G_;
    float a = bg[g];
    for (int k = 0; k < H_; ++k) a += Wg[g * H_ + k] * ctx[(size_t)n * H_ + k];
    outg[idx] = a;
}

// ---- out[n,t,:] = sw[n] * softmax_V(logits[n]). grid = NB_, block = 256 ----
__global__ void k_vocab_final(const float* __restrict__ logits,
                              const float* __restrict__ sw,
                              float* __restrict__ out, int t) {
    __shared__ float red[256];
    int n = blockIdx.x, tid = threadIdx.x;
    const float* lr = logits + (size_t)n * V_;
    float m = -INFINITY;
    for (int v = tid; v < V_; v += 256) m = fmaxf(m, lr[v]);
    red[tid] = m; __syncthreads();
    for (int off = 128; off > 0; off >>= 1) {
        if (tid < off) red[tid] = fmaxf(red[tid], red[tid + off]);
        __syncthreads();
    }
    m = red[0]; __syncthreads();
    float ssum = 0.0f;
    for (int v = tid; v < V_; v += 256) ssum += __expf(lr[v] - m);
    red[tid] = ssum; __syncthreads();
    for (int off = 128; off > 0; off >>= 1) {
        if (tid < off) red[tid] += red[tid + off];
        __syncthreads();
    }
    float scale = sw[n] / red[0];
    float* orow = out + ((size_t)n * T_ + t) * V_;
    for (int v = tid; v < V_; v += 256) orow[v] = __expf(lr[v] - m) * scale;
}

// ---- pointer scatter ----
__global__ void k_scatter(const float* __restrict__ prob, const float* __restrict__ sw,
                          const int* __restrict__ story, float* __restrict__ out, int t) {
    int idx = blockIdx.x * blockDim.x + threadIdx.x;
    if (idx >= NB_ * S_) return;
    int n = idx / S_, s = idx % S_, b = n % B_;
    float w = (1.0f - sw[n]) * prob[idx];
    int tok = story[b * S_ + s];
    atomicAdd(out + ((size_t)n * T_ + t) * V_ + tok, w);
}

// ============================================================================
extern "C" void kernel_launch(void* const* d_in, const int* in_sizes, int n_in,
                              void* d_out, int out_size, void* d_ws, size_t ws_size,
                              hipStream_t stream) {
    (void)in_sizes; (void)n_in; (void)out_size; (void)ws_size;
    const int*   story  = (const int*)d_in[0];
    const int*   lens   = (const int*)d_in[1];
    const int*   tgt    = (const int*)d_in[2];
    const int*   dom    = (const int*)d_in[3];
    const int*   slo    = (const int*)d_in[4];
    const float* emb    = (const float*)d_in[5];
    const float* Wih_f  = (const float*)d_in[6];
    const float* Whh_f  = (const float*)d_in[7];
    const float* bih_f  = (const float*)d_in[8];
    const float* bhh_f  = (const float*)d_in[9];
    const float* Wih_b  = (const float*)d_in[10];
    const float* Whh_b  = (const float*)d_in[11];
    const float* bih_b  = (const float*)d_in[12];
    const float* bhh_b  = (const float*)d_in[13];
    const float* dWih   = (const float*)d_in[14];
    const float* dWhh   = (const float*)d_in[15];
    const float* dbih   = (const float*)d_in[16];
    const float* dbhh   = (const float*)d_in[17];
    const float* Wr     = (const float*)d_in[18];
    const float* br     = (const float*)d_in[19];
    const float* Wg     = (const float*)d_in[20];
    const float* bg     = (const float*)d_in[21];
    const float* slot_t = (const float*)d_in[22];
    float* out = (float*)d_out;
    float* out_gates = out + (size_t)NSLOT_ * B_ * T_ * V_;

    // ---- workspace carve-up ----
    char* ws = (char*)d_ws;
    size_t off = 0;
    auto alloc = [&](size_t bytes) -> char* {
        char* p = ws + off;
        off = (off + bytes + 255) & ~(size_t)255;
        return p;
    };
    __bf16* emb_bf   = (__bf16*)alloc((size_t)V_ * KP_ * 2);
    __bf16* Wihf_bf  = (__bf16*)alloc((size_t)H3_ * KP_ * 2);
    __bf16* Whhf_bf  = (__bf16*)alloc((size_t)H3_ * KP_ * 2);
    __bf16* Wihb_bf  = (__bf16*)alloc((size_t)H3_ * KP_ * 2);
    __bf16* Whhb_bf  = (__bf16*)alloc((size_t)H3_ * KP_ * 2);
    __bf16* dWih_bf  = (__bf16*)alloc((size_t)H3_ * KP_ * 2);
    __bf16* dWhh_bf  = (__bf16*)alloc((size_t)H3_ * KP_ * 2);
    __bf16* x_bf     = (__bf16*)alloc((size_t)S_ * B_ * KP_ * 2);
    float*  gi_f     = (float*) alloc((size_t)S_ * B_ * H3_ * 4);
    float*  gi_b     = (float*) alloc((size_t)S_ * B_ * H3_ * 4);
    float*  enc_acc  = (float*) alloc((size_t)B_ * S_ * H_ * 4);
    float*  h_fb     = (float*) alloc((size_t)2 * B_ * H_ * 4);
    float*  ench     = (float*) alloc((size_t)B_ * H_ * 4);
    float*  decin_f  = (float*) alloc((size_t)T_ * NB_ * H_ * 4);
    __bf16* decin_bf = (__bf16*)alloc((size_t)T_ * NB_ * KP_ * 2);
    float*  gi_dec   = (float*) alloc((size_t)T_ * NB_ * H3_ * 4);
    float*  h_dec    = (float*) alloc((size_t)NB_ * H_ * 4);
    __bf16* hdec_bf  = (__bf16*)alloc((size_t)NB_ * KP_ * 2);
    float*  gh       = (float*) alloc((size_t)NB_ * H3_ * 4);
    float*  prob     = (float*) alloc((size_t)NB_ * S_ * 4);
    float*  ctx      = (float*) alloc((size_t)NB_ * H_ * 4);
    float*  swv      = (float*) alloc((size_t)NB_ * 4);
    float*  logits   = (float*) alloc((size_t)NB_ * V_ * 4);

    const int TB = 256;
    auto g1 = [](size_t n) { return dim3((unsigned)((n + 255) / 256)); };

    // ---- zero encoder output accumulator ----
    hipMemsetAsync(enc_acc, 0, (size_t)B_ * S_ * H_ * 4, stream);

    // ---- casts to padded bf16 ----
    k_cast_pad<<<g1((size_t)V_ * KP_),  TB, 0, stream>>>(emb,   emb_bf,  V_);
    k_cast_pad<<<g1((size_t)H3_ * KP_), TB, 0, stream>>>(Wih_f, Wihf_bf, H3_);
    k_cast_pad<<<g1((size_t)H3_ * KP_), TB, 0, stream>>>(Whh_f, Whhf_bf, H3_);
    k_cast_pad<<<g1((size_t)H3_ * KP_), TB, 0, stream>>>(Wih_b, Wihb_bf, H3_);
    k_cast_pad<<<g1((size_t)H3_ * KP_), TB, 0, stream>>>(Whh_b, Whhb_bf, H3_);
    k_cast_pad<<<g1((size_t)H3_ * KP_), TB, 0, stream>>>(dWih,  dWih_bf, H3_);
    k_cast_pad<<<g1((size_t)H3_ * KP_), TB, 0, stream>>>(dWhh,  dWhh_bf, H3_);

    // ---- embeddings ----
    k_embed_story<<<g1((size_t)S_ * B_ * KP_), TB, 0, stream>>>(story, emb, x_bf);
    k_dec_embed<<<g1((size_t)T_ * NB_ * KP_), TB, 0, stream>>>(
        tgt, dom, slo, emb, slot_t, decin_f, decin_bf);

    // ---- hoisted input-side GEMMs: gi = x @ Wih^T + bih (WMMA bf16) ----
    k_gemm_bf16<3><<<dim3(H3_ / 48, (S_ * B_) / 16), 32, 0, stream>>>(
        x_bf, KP_, Wihf_bf, KP_, gi_f, H3_, bih_f);
    k_gemm_bf16<3><<<dim3(H3_ / 48, (S_ * B_) / 16), 32, 0, stream>>>(
        x_bf, KP_, Wihb_bf, KP_, gi_b, H3_, bih_b);
    k_gemm_bf16<3><<<dim3(H3_ / 48, (T_ * NB_) / 16), 32, 0, stream>>>(
        decin_bf, KP_, dWih_bf, KP_, gi_dec, H3_, dbih);

    // ---- fused bi-directional encoder scan: ONE launch for all 512 steps ----
    k_enc_scan<<<2, 1024, 0, stream>>>(gi_f, gi_b, Whhf_bf, Whhb_bf,
                                       bhh_f, bhh_b, lens, enc_acc, h_fb);

    // ---- enc_hidden = h_f + h_b; decoder hidden init ----
    k_add<<<g1((size_t)B_ * H_), TB, 0, stream>>>(h_fb, h_fb + B_ * H_, ench, B_ * H_);
    k_hdec_init<<<g1((size_t)NB_ * KP_), TB, 0, stream>>>(ench, h_dec, hdec_bf);

    // ---- decoder ----
    for (int t = 0; t < T_; ++t) {
        k_gemm_bf16<3><<<dim3(H3_ / 48, NB_ / 16), 32, 0, stream>>>(
            hdec_bf, KP_, dWhh_bf, KP_, gh, H3_, dbhh);
        k_dec_gate<<<g1((size_t)NB_ * H_), TB, 0, stream>>>(gi_dec, gh, h_dec, hdec_bf, t);

        k_scores<<<g1((size_t)NB_ * S_), TB, 0, stream>>>(h_dec, enc_acc, prob);
        k_attn_softmax<<<NB_, S_, 0, stream>>>(prob, lens);
        k_context<<<NB_, TB, 0, stream>>>(prob, enc_acc, ctx);
        k_sw<<<NB_, TB, 0, stream>>>(h_dec, ctx, decin_f, Wr, br, swv, t);
        if (t == 0)
            k_gates<<<g1((size_t)NB_ * G_), TB, 0, stream>>>(ctx, Wg, bg, out_gates);

        // vocab logits: (480 x 416) @ (20000 x 416)^T -> WMMA bf16, B streams L2
        k_gemm_bf16<5><<<dim3(V_ / 80, NB_ / 16), 32, 0, stream>>>(
            hdec_bf, KP_, emb_bf, KP_, logits, V_, nullptr);

        k_vocab_final<<<NB_, TB, 0, stream>>>(logits, swv, out, t);
        k_scatter<<<g1((size_t)NB_ * S_), TB, 0, stream>>>(prob, swv, story, out, t);
    }
}
